// HybridFCLQuantum_65481071403652
// MI455X (gfx1250) — matrix operand order repaired and checked by reading, not compiled, not run on hardware
//
#include <hip/hip_runtime.h>
#include <hip/hip_bf16.h>
#include <math.h>

#define NQ    12
#define DIM   4096      // 2^NQ
#define BATCH 512
#define DEPTH 4
#define NPAD  16        // W padded to 16 rows so B loads are unconditional

typedef __attribute__((ext_vector_type(16))) _Float16 v16h;
typedef __attribute__((ext_vector_type(8)))  _Float16 v8h;
typedef __attribute__((ext_vector_type(8)))  float    v8f;

// ---------------------------------------------------------------------------
// Kernel 0: fold the linear head into the SIGNS matrix, padded to 16 rows.
//   W[c][k] = sum_i head_w[c,i] * (1 - 2*((k >> (NQ-1-i)) & 1)),  c in {0,1}
//   W[c][k] = 0 for c in 2..15 (padding -> unconditional WMMA B loads)
// ---------------------------------------------------------------------------
__global__ void build_w_kernel(const float* __restrict__ head_w,
                               _Float16* __restrict__ Wbuf) {
    int k = blockIdx.x * blockDim.x + threadIdx.x;
    if (k >= DIM) return;
    float w0 = 0.f, w1 = 0.f;
    #pragma unroll
    for (int i = 0; i < NQ; ++i) {
        float sg = ((k >> (NQ - 1 - i)) & 1) ? -1.f : 1.f;
        w0 += head_w[i]      * sg;
        w1 += head_w[NQ + i] * sg;
    }
    Wbuf[k]       = (_Float16)w0;
    Wbuf[DIM + k] = (_Float16)w1;
    #pragma unroll
    for (int c = 2; c < NPAD; ++c)
        Wbuf[(size_t)c * DIM + k] = (_Float16)0.0f;
}

// ---------------------------------------------------------------------------
// Kernel 1: state-vector simulation, one sample per workgroup, state in LDS.
// Real gates => simulate real/imag parts u,v independently.
// Writes probs (scaled by 256 to keep f16 in the normal range) to scratch.
// ---------------------------------------------------------------------------
__global__ void __launch_bounds__(256)
sim_kernel(const float* __restrict__ x, const float* __restrict__ params,
           _Float16* __restrict__ probs) {
    __shared__ float u[DIM];
    __shared__ float v[DIM];
    __shared__ float cb[NQ], sb[NQ];
    __shared__ float cy[DEPTH * NQ], sy[DEPTH * NQ];

    const int b = blockIdx.x;
    const int t = threadIdx.x;

    if (t < NQ) {                       // RX half-angles for this sample
        float a = 0.5f * x[b * NQ + t];
        cb[t] = cosf(a);
        sb[t] = sinf(a);
    }
    if (t < DEPTH * NQ) {               // RY half-angles (diagonal of params)
        int d = t / NQ, i = t % NQ;
        float th = 0.5f * params[(d * NQ + i) * NQ + i];
        cy[t] = cosf(th);
        sy[t] = sinf(th);
    }
    __syncthreads();

    // Post-RX product state: amp(k) = (-i)^popcount(k) * prod(c_i or s_i)
    for (int idx = t; idx < DIM; idx += 256) {
        float r = 1.f;
        #pragma unroll
        for (int i = 0; i < NQ; ++i)
            r *= ((idx >> (NQ - 1 - i)) & 1) ? sb[i] : cb[i];
        int p = __popc(idx) & 3;
        u[idx] = (p == 0) ? r : ((p == 2) ? -r : 0.f);
        v[idx] = (p == 1) ? -r : ((p == 3) ? r : 0.f);
    }
    __syncthreads();

    for (int d = 0; d < DEPTH; ++d) {
        // 12 RY butterflies (real rotation applied to u and v)
        for (int i = 0; i < NQ; ++i) {
            const float c = cy[d * NQ + i], s = sy[d * NQ + i];
            const int m = NQ - 1 - i;           // bit index of wire i
            const int lomask = (1 << m) - 1;
            for (int j = t; j < DIM / 2; j += 256) {
                int i0 = ((j >> m) << (m + 1)) | (j & lomask);
                int i1 = i0 | (1 << m);
                float u0 = u[i0], u1 = u[i1];
                u[i0] = c * u0 - s * u1;
                u[i1] = s * u0 + c * u1;
                float v0 = v[i0], v1 = v[i1];
                v[i0] = c * v0 - s * v1;
                v[i1] = s * v0 + c * v1;
            }
            __syncthreads();
        }
        // CZ chain fused into one sign pass: parity of adjacent 1-1 bit pairs
        for (int idx = t; idx < DIM; idx += 256) {
            if (__popc(idx & (idx >> 1)) & 1) {
                u[idx] = -u[idx];
                v[idx] = -v[idx];
            }
        }
        __syncthreads();
    }

    for (int idx = t; idx < DIM; idx += 256) {
        float p = u[idx] * u[idx] + v[idx] * v[idx];
        probs[(size_t)b * DIM + idx] = (_Float16)(p * 256.0f);
    }
}

// ---------------------------------------------------------------------------
// Kernel 2: logits = (probs/256) @ W^T + head_b via v_wmma_f32_16x16x32_f16,
// then 2-class log-softmax. One wave per 16-row batch tile (32 tiles).
//
// WMMA wave32 layouts (ISA 7.12.2):
//  A 16x32 f16 : lane l -> M = l&15 ; elems 0-7 : K = kk + 8*(l>>4) + e
//                                     elems 8-15: K = kk + 16 + 8*(l>>4) + (e-8)
//  B 32x16 f16 : lane l -> N = l&15 ; elems    : same K order as A (kk + 8*(l>>4) + e, +16 for 8-15)
//  D 16x16 f32 : lane l -> N = l&15 ; acc[r]   : M = r + 8*(l>>4)
// ---------------------------------------------------------------------------
__global__ void __launch_bounds__(32)
head_wmma_kernel(const _Float16* __restrict__ probs,
                 const _Float16* __restrict__ Wbuf,
                 const float* __restrict__ head_b,
                 float* __restrict__ out) {
    const int tile = blockIdx.x;        // 16 batch rows per tile
    const int lane = threadIdx.x;       // wave32
    const int nm = lane & 15;
    const int hi = lane >> 4;

    v8f acc = {};
    const _Float16* arow = probs + (size_t)(tile * 16 + nm) * DIM;
    const _Float16* brow = Wbuf + (size_t)nm * DIM;   // rows 2..15 are zeros

    for (int kk = 0; kk < DIM; kk += 32) {
        // Prefetch next A chunk (one 512B stride ahead) -> global_prefetch_b8
        __builtin_prefetch(arow + kk + 256, 0, 0);

        // A fragment: two contiguous 16B loads per lane
        v8h a0 = *(const v8h*)(arow + kk + hi * 8);
        v8h a1 = *(const v8h*)(arow + kk + hi * 8 + 16);
        v16h A = __builtin_shufflevector(a0, a1,
                 0, 1, 2, 3, 4, 5, 6, 7, 8, 9, 10, 11, 12, 13, 14, 15);
        // B fragment: unconditional (padded rows are zero)
        v8h b0 = *(const v8h*)(brow + kk + hi * 8);
        v8h b1 = *(const v8h*)(brow + kk + hi * 8 + 16);
        v16h B = __builtin_shufflevector(b0, b1,
                 0, 1, 2, 3, 4, 5, 6, 7, 8, 9, 10, 11, 12, 13, 14, 15);

        acc = __builtin_amdgcn_wmma_f32_16x16x32_f16(
                  /*neg_a=*/false, A, /*neg_b=*/false, B,
                  /*c_mod=*/(short)0, acc,
                  /*reuse_a=*/false, /*reuse_b=*/false);
    }

    const float bias = (nm < 2) ? head_b[nm] : 0.f;
    #pragma unroll
    for (int r = 0; r < 8; ++r) {
        float la = acc[r] * (1.0f / 256.0f) + bias;  // own-class logit
        float lb = __shfl_xor(la, 1, 32);            // other-class logit (n^1)
        if (nm < 2) {
            float mx  = fmaxf(la, lb);
            float lse = mx + logf(expf(la - mx) + expf(lb - mx));
            int row = tile * 16 + hi * 8 + r;
            out[row * 2 + nm] = la - lse;
        }
    }
}

// ---------------------------------------------------------------------------
extern "C" void kernel_launch(void* const* d_in, const int* in_sizes, int n_in,
                              void* d_out, int out_size, void* d_ws, size_t ws_size,
                              hipStream_t stream) {
    const float* x       = (const float*)d_in[0];   // (512, 12)
    const float* params  = (const float*)d_in[1];   // (48, 12)
    const float* head_w  = (const float*)d_in[2];   // (2, 12)
    const float* head_b  = (const float*)d_in[3];   // (2,)
    float* out = (float*)d_out;                     // (512, 2) f32

    // scratch layout: [ probs f16 512*4096 | W f16 16*4096 ]  (~4.15 MB)
    _Float16* probs = (_Float16*)d_ws;
    _Float16* Wbuf  = probs + (size_t)BATCH * DIM;

    build_w_kernel<<<(DIM + 255) / 256, 256, 0, stream>>>(head_w, Wbuf);
    sim_kernel<<<BATCH, 256, 0, stream>>>(x, params, probs);
    head_wmma_kernel<<<BATCH / 16, 32, 0, stream>>>(probs, Wbuf, head_b, out);
}
